// GNNModel_52450140618893
// MI455X (gfx1250) — matrix operand order repaired and checked by reading, not compiled, not run on hardware
//
#include <hip/hip_runtime.h>
#include <hip/hip_bf16.h>

typedef __attribute__((ext_vector_type(16))) __bf16 v16bf;
typedef __attribute__((ext_vector_type(4)))  __bf16 v4bf;
typedef __attribute__((ext_vector_type(8)))  float  v8f;

#define HC 256   // hidden channels

// LDS byte offset of a __shared__ object: generic flat address low 32 bits are
// the wave-relative LDS address (ISA 10.2 aperture mapping).
__device__ __forceinline__ unsigned lds_off(const void* p) {
    return (unsigned)(unsigned long long)p;
}

__device__ __forceinline__ void async_b128(unsigned ldsByte, const void* g) {
    asm volatile("global_load_async_to_lds_b128 %0, %1, off"
                 :: "v"(ldsByte), "v"((unsigned long long)g) : "memory");
}
__device__ __forceinline__ void wait_async0() {
    asm volatile("s_wait_asynccnt 0x0" ::: "memory");
}

// ---------------------------------------------------------------------------
// WMMA bf16 GEMM: C[M,N] = act(A @ B + bias), A bf16 [M,K], Bt bf16 [N,K]
// (weights pre-transposed). Block = 256 threads (8 waves) computing a
// 64 x NCOLS tile: 4 row strips x 2 column halves, NCOLS/32 WMMAs per wave
// per 32-deep k-step. Tiles staged via global_load_async_to_lds_b128.
// Requires K % 32 == 0, N % NCOLS == 0.
// ---------------------------------------------------------------------------
template <int NCOLS>
__global__ __launch_bounds__(256)
void gemm_bf16(const __bf16* __restrict__ A, const __bf16* __restrict__ Bt,
               const float* __restrict__ bias,
               float* __restrict__ C32, __bf16* __restrict__ C16,
               int M, int K, int N, int doRelu)
{
    constexpr int CW = NCOLS / 32;            // col tiles per wave
    __shared__ __bf16 As[64 * 32];            // [row][k]
    __shared__ __bf16 Bs[NCOLS * 32];         // [n][k]

    const int tid      = threadIdx.x;
    const int lane     = tid & 31;
    const int wave     = tid >> 5;
    const int rowStrip = wave & 3;            // 0..3
    const int colHalf  = wave >> 2;           // 0..1
    const int rowBlk   = blockIdx.y * 64;
    const int n0       = blockIdx.x * NCOLS;

    const int  ar    = tid >> 2;              // A chunk row (64 rows x 4 chunks)
    const int  ac    = tid & 3;
    const bool fullA = (rowBlk + 64 <= M);

    v8f acc[CW] = {};

    for (int k0 = 0; k0 < K; k0 += 32) {
        // ---- stage A tile 64x32 bf16 (1 x 16B chunk per thread) ----
        if (fullA) {
            async_b128(lds_off(&As[ar * 32 + ac * 8]),
                       A + (size_t)(rowBlk + ar) * K + k0 + ac * 8);
        } else {
            uint4 z = make_uint4(0u, 0u, 0u, 0u);
            int gr = rowBlk + ar;
            if (gr < M) z = *(const uint4*)(A + (size_t)gr * K + k0 + ac * 8);
            *(uint4*)&As[ar * 32 + ac * 8] = z;
        }
        // ---- stage B tile NCOLS x 32 bf16 (already [n][k] in global) ----
        #pragma unroll
        for (int i = 0; i < NCOLS / 64; ++i) {
            int e  = tid + i * 256;           // chunk id, NCOLS*4 total
            int bn = e >> 2;
            int bc = e & 3;
            async_b128(lds_off(&Bs[bn * 32 + bc * 8]),
                       Bt + (size_t)(n0 + bn) * K + k0 + bc * 8);
        }
        wait_async0();
        __syncthreads();

        // ---- A fragment (ISA 16-bit A 16x32 layout) ----
        const int m  = lane & 15;
        const int kb = (lane < 16) ? 0 : 8;
        const __bf16* arow = &As[(rowStrip * 16 + m) * 32 + kb];
        v16bf a;
        #pragma unroll
        for (int i = 0; i < 8; ++i) { a[i] = arow[i]; a[8 + i] = arow[16 + i]; }

        // ---- CW col tiles, B fragment column-striped ----
        const int koff = (lane < 16) ? 0 : 8;
        #pragma unroll
        for (int c = 0; c < CW; ++c) {
            const __bf16* bcol = &Bs[(colHalf * (NCOLS / 2) + c * 16 + m) * 32 + koff];
            v16bf b;
            #pragma unroll
            for (int i = 0; i < 8; ++i) { b[i] = bcol[i]; b[8 + i] = bcol[16 + i]; }
            acc[c] = __builtin_amdgcn_wmma_f32_16x16x32_bf16(
                false, a, false, b, (short)0, acc[c], false, false);
        }
        __syncthreads();
    }

    // ---- epilogue: bias + optional relu; fp32 or bf16 store ----
    const int n     = lane & 15;
    const int rhalf = (lane < 16) ? 0 : 8;
    #pragma unroll
    for (int c = 0; c < CW; ++c) {
        int col  = n0 + colHalf * (NCOLS / 2) + c * 16 + n;
        float bv = bias ? bias[col] : 0.0f;
        #pragma unroll
        for (int i = 0; i < 8; ++i) {
            int gr = rowBlk + rowStrip * 16 + rhalf + i;
            if (gr < M) {
                float v = acc[c][i] + bv;
                if (doRelu) v = fmaxf(v, 0.0f);
                if (C16) C16[(size_t)gr * N + col] = (__bf16)v;
                else     C32[(size_t)gr * N + col] = v;
            }
        }
    }
}

// ---------------------------------------------------------------------------
// Conversion / elementwise / scatter kernels
// ---------------------------------------------------------------------------
__global__ void cvt_bf16_4(const float* __restrict__ s, __bf16* __restrict__ d, int n4) {
    int i = blockIdx.x * blockDim.x + threadIdx.x;
    if (i >= n4) return;
    float4 v = ((const float4*)s)[i];
    v4bf o;
    o[0] = (__bf16)v.x; o[1] = (__bf16)v.y; o[2] = (__bf16)v.z; o[3] = (__bf16)v.w;
    ((v4bf*)d)[i] = o;
}

// Wt[n*K+k] = bf16(W[k*N+n])
__global__ void wtrans(const float* __restrict__ W, __bf16* __restrict__ Wt, int K, int N) {
    int i = blockIdx.x * blockDim.x + threadIdx.x;
    if (i >= K * N) return;
    int k = i / N, n = i - k * N;
    Wt[(size_t)n * K + k] = (__bf16)W[i];
}

__global__ void fill_f32(float* p, float v, int n) {
    int i = blockIdx.x * blockDim.x + threadIdx.x;
    if (i < n) p[i] = v;
}

__global__ void deg_accum(const int* __restrict__ dst, float* __restrict__ deg, int E) {
    int i = blockIdx.x * blockDim.x + threadIdx.x;
    if (i < E) atomicAdd(&deg[dst[i]], 1.0f);
}

__global__ void rsqrt_inplace(float* p, int n) {
    int i = blockIdx.x * blockDim.x + threadIdx.x;
    if (i < n) p[i] = rsqrtf(p[i]);
}

// agg[i] = m[i] * dinv[node]^2  (self loop); n4 = N*64 float4s
__global__ void self_init(const float* __restrict__ m, const float* __restrict__ dinv,
                          float* __restrict__ agg, int n4) {
    int i = blockIdx.x * blockDim.x + threadIdx.x;
    if (i >= n4) return;
    int node = i >> 6;
    float w = dinv[node]; w *= w;
    float4 v = ((const float4*)m)[i];
    v.x *= w; v.y *= w; v.z *= w; v.w *= w;
    ((float4*)agg)[i] = v;
}

// one wave per edge; lane covers 2 float4 chunks of 256 channels
__global__ __launch_bounds__(256)
void edge_scatter(const int* __restrict__ src, const int* __restrict__ dst,
                  const float* __restrict__ m, const float* __restrict__ dinv,
                  float* __restrict__ agg, int E)
{
    int e    = blockIdx.x * 8 + (threadIdx.x >> 5);
    int lane = threadIdx.x & 31;
    if (e >= E) return;
    int s = src[e], d = dst[e];
    float w = dinv[s] * dinv[d];
    const float* ms = m   + (size_t)s * HC;
    float*       ad = agg + (size_t)d * HC;
    #pragma unroll
    for (int i = 0; i < 2; ++i) {
        int c4 = lane + i * 32;
        float4 v = *(const float4*)(ms + c4 * 4);
        atomicAdd(ad + c4 * 4 + 0, v.x * w);
        atomicAdd(ad + c4 * 4 + 1, v.y * w);
        atomicAdd(ad + c4 * 4 + 2, v.z * w);
        atomicAdd(ad + c4 * 4 + 3, v.w * w);
    }
}

// h16 = bf16(relu((agg + b_g - mean)*gamma*rsqrt(var+eps) + beta))
__global__ void bn_relu_bf16(const float* __restrict__ agg, __bf16* __restrict__ h,
                             const float* __restrict__ bg, const float* __restrict__ gamma,
                             const float* __restrict__ beta, const float* __restrict__ mean,
                             const float* __restrict__ var, int total)
{
    int i = blockIdx.x * blockDim.x + threadIdx.x;
    if (i >= total) return;
    int c = i & (HC - 1);
    float scale = gamma[c] * rsqrtf(var[c] + 1e-5f);
    float v = (agg[i] + bg[c] - mean[c]) * scale + beta[c];
    h[i] = (__bf16)fmaxf(v, 0.0f);
}

__global__ void pool_sum(const __bf16* __restrict__ h, const int* __restrict__ batch,
                         float* __restrict__ sums, int total) {
    int i = blockIdx.x * blockDim.x + threadIdx.x;
    if (i >= total) return;
    int node = i >> 8;
    int c    = i & (HC - 1);
    atomicAdd(&sums[batch[node] * HC + c], (float)h[i]);
}

__global__ void pool_cnt(const int* __restrict__ batch, float* __restrict__ cnts, int N) {
    int i = blockIdx.x * blockDim.x + threadIdx.x;
    if (i < N) atomicAdd(&cnts[batch[i]], 1.0f);
}

__global__ void pool_div(const float* __restrict__ sums, const float* __restrict__ cnts,
                         __bf16* __restrict__ g, int n) {
    int i = blockIdx.x * blockDim.x + threadIdx.x;
    if (i >= n) return;
    g[i] = (__bf16)(sums[i] / fmaxf(cnts[i >> 8], 1.0f));
}

// ---------------------------------------------------------------------------
extern "C" void kernel_launch(void* const* d_in, const int* in_sizes, int n_in,
                              void* d_out, int out_size, void* d_ws, size_t ws_size,
                              hipStream_t stream)
{
    const float* x     = (const float*)d_in[0];
    const int*   edge  = (const int*)  d_in[1];   // [2, E]
    const int*   batch = (const int*)  d_in[2];
    const float* W_in  = (const float*)d_in[3];
    const float* b_in  = (const float*)d_in[4];
    const float* W_g   = (const float*)d_in[5];   // [3,256,256]
    const float* b_g   = (const float*)d_in[6];
    const float* gam   = (const float*)d_in[7];
    const float* bet   = (const float*)d_in[8];
    const float* mea   = (const float*)d_in[9];
    const float* var   = (const float*)d_in[10];
    const float* W1    = (const float*)d_in[11];
    const float* b1    = (const float*)d_in[12];
    const float* W2    = (const float*)d_in[13];
    const float* b2    = (const float*)d_in[14];

    const int N = in_sizes[2];          // 100000
    const int E = in_sizes[1] / 2;      // 1600000
    const int F = in_sizes[0] / N;      // 768
    const int G = 64, O = 128;

    const int* e_src = edge;
    const int* e_dst = edge + E;

    // ---- workspace layout (regions 256B-aligned) ----
    float* fws = (float*)d_ws;
    size_t of = 0;
    auto allocF = [&](size_t n) { float* p = fws + of; of += (n + 63) & ~(size_t)63; return p; };
    float* hA   = allocF((size_t)N * HC);       // agg
    float* mB   = allocF((size_t)N * HC);       // GEMM output m
    float* deg  = allocF(N);                    // then dinv
    float* sums = allocF((size_t)G * HC);
    float* cnts = allocF(G);

    __bf16* bws = (__bf16*)(fws + of);
    size_t ob = 0;
    auto allocH = [&](size_t n) { __bf16* p = bws + ob; ob += (n + 127) & ~(size_t)127; return p; };
    __bf16* x16  = allocH((size_t)N * F);
    __bf16* h16  = allocH((size_t)N * HC);
    __bf16* g16  = allocH((size_t)G * HC);
    __bf16* t16  = allocH((size_t)G * O);
    __bf16* WinT = allocH((size_t)F * HC);      // [HC][F]
    __bf16* WgT  = allocH((size_t)3 * HC * HC); // [HC][HC] x3
    __bf16* W1T  = allocH((size_t)HC * O);      // [O][HC]
    __bf16* W2T  = allocH((size_t)O * O);       // [O][O]

    const int TB = 256;
    auto cdiv = [](int a, int b) { return (a + b - 1) / b; };

    // ---- precision prep: activations bf16, weights bf16-transposed ----
    cvt_bf16_4<<<cdiv(N * F / 4, TB), TB, 0, stream>>>(x, x16, N * F / 4);
    wtrans<<<cdiv(F * HC, TB), TB, 0, stream>>>(W_in, WinT, F, HC);
    for (int l = 0; l < 3; ++l)
        wtrans<<<cdiv(HC * HC, TB), TB, 0, stream>>>(W_g + (size_t)l * HC * HC,
                                                     WgT + (size_t)l * HC * HC, HC, HC);
    wtrans<<<cdiv(HC * O, TB), TB, 0, stream>>>(W1, W1T, HC, O);
    wtrans<<<cdiv(O * O, TB), TB, 0, stream>>>(W2, W2T, O, O);

    // ---- degree (1 self loop + in-edges) -> dinv ----
    fill_f32<<<cdiv(N, TB), TB, 0, stream>>>(deg, 1.0f, N);
    deg_accum<<<cdiv(E, TB), TB, 0, stream>>>(e_dst, deg, E);
    rsqrt_inplace<<<cdiv(N, TB), TB, 0, stream>>>(deg, N);

    // ---- input projection: h = relu(x @ W_in + b_in), bf16 out ----
    {
        dim3 grid(HC / 256, cdiv(N, 64));
        gemm_bf16<256><<<grid, 256, 0, stream>>>(x16, WinT, b_in, nullptr, h16,
                                                 N, F, HC, 1);
    }

    // ---- 3 GCN layers ----
    for (int l = 0; l < 3; ++l) {
        dim3 grid(HC / 256, cdiv(N, 64));
        gemm_bf16<256><<<grid, 256, 0, stream>>>(h16, WgT + (size_t)l * HC * HC,
                                                 nullptr, mB, nullptr, N, HC, HC, 0);
        self_init<<<cdiv(N * (HC / 4), TB), TB, 0, stream>>>(mB, deg, hA, N * (HC / 4));
        edge_scatter<<<cdiv(E, 8), TB, 0, stream>>>(e_src, e_dst, mB, deg, hA, E);
        bn_relu_bf16<<<cdiv(N * HC, TB), TB, 0, stream>>>(hA, h16,
                b_g + l * HC, gam + l * HC, bet + l * HC,
                mea + l * HC, var + l * HC, N * HC);
    }

    // ---- global mean pool ----
    fill_f32<<<cdiv(G * HC + G, TB), TB, 0, stream>>>(sums, 0.0f, G * HC + G);
    pool_sum<<<cdiv(N * HC, TB), TB, 0, stream>>>(h16, batch, sums, N * HC);
    pool_cnt<<<cdiv(N, TB), TB, 0, stream>>>(batch, cnts, N);
    pool_div<<<cdiv(G * HC, TB), TB, 0, stream>>>(sums, cnts, g16, G * HC);

    // ---- output MLP: out = relu(g @ W1 + b1) @ W2 + b2 ----
    {
        dim3 grid1(O / 128, cdiv(G, 64));
        gemm_bf16<128><<<grid1, 256, 0, stream>>>(g16, W1T, b1, nullptr, t16, G, HC, O, 1);
        dim3 grid2(O / 128, cdiv(G, 64));
        gemm_bf16<128><<<grid2, 256, 0, stream>>>(t16, W2T, b2, (float*)d_out, nullptr,
                                                  G, O, O, 0);
    }
}
